// SparseMPNN_8126078124639
// MI455X (gfx1250) — matrix operand order repaired and checked by reading, not compiled, not run on hardware
//
#include <hip/hip_runtime.h>
#include <hip/hip_bf16.h>

// GatedGraphConv (L=1) for MI455X / gfx1250, wave32 + v_wmma_f32_16x16x32_bf16.
// Pipeline: prep(convert/zero) -> GEMM1 (m = x@W, bf16 WMMA, 2x2 reg-blocked)
// -> edge scatter (bf16 gather, f32 atomics into L2-resident agg) ->
// cvt(agg->bf16) -> fused GRU (2-row reg-blocked gi/gh WMMA + gating epilogue).

typedef __attribute__((ext_vector_type(16))) __bf16 v16bf;
typedef __attribute__((ext_vector_type(8)))  float  v8f;

constexpr int B = 8, N = 4096, D = 256, E = 65536;
constexpr int BN = B * N;

union FragAB { uint4 q[2]; v16bf v; };
union Pack8  { uint4 q; __bf16 h[8]; };

__device__ __forceinline__ v8f wmma_bf16(v16bf a, v16bf b, v8f c) {
  return __builtin_amdgcn_wmma_f32_16x16x32_bf16(false, a, false, b, (short)0, c,
                                                 false, false);
}

// A-matrix 16x32 bf16 frag (row-major source, row stride ldA elements).
// Lane L: row = row0 + (L&15); elems 0..7 -> K = k0 + hs*8 + e,
//         elems 8..15 -> K = k0 + 16 + hs*8 + (e-8), hs = L>>4.
__device__ __forceinline__ v16bf load_a_frag(const __bf16* A, int ldA,
                                             int row0, int k0, int lane) {
  const int hs = lane >> 4;
  const int mr = lane & 15;
  const __bf16* p = A + (size_t)(row0 + mr) * ldA + k0 + hs * 8;
  FragAB f;
  f.q[0] = *reinterpret_cast<const uint4*>(p);
  f.q[1] = *reinterpret_cast<const uint4*>(p + 16);
  return f.v;
}

// B-matrix 32x16 bf16 frag from K-major storage Bt[n][k] (row stride ldB).
// Lane L: col n = n0 + (L&15); elems 0..7 -> K = k0 + hs*8 + e,
//         elems 8..15 -> K = k0 + 16 + hs*8 + (e-8), hs = L>>4.
__device__ __forceinline__ v16bf load_b_frag(const __bf16* Bt, int ldB,
                                             int n0, int k0, int lane) {
  const int hs = lane >> 4;
  const int nc = lane & 15;
  const __bf16* p = Bt + (size_t)(n0 + nc) * ldB + k0 + hs * 8;
  FragAB f;
  f.q[0] = *reinterpret_cast<const uint4*>(p);
  f.q[1] = *reinterpret_cast<const uint4*>(p + 16);
  return f.v;
}

__device__ __forceinline__ float sigmoidf(float v) {
  return 1.0f / (1.0f + expf(-v));
}

// ---- prep: x -> bf16, zero agg -------------------------------------------
__global__ void k_prep_nodes(const float* __restrict__ x, __bf16* __restrict__ xh,
                             float* __restrict__ agg, int n) {
  int i = blockIdx.x * blockDim.x + threadIdx.x;
  if (i < n) { xh[i] = (__bf16)x[i]; agg[i] = 0.0f; }
}

// ---- prep: weights -> bf16 (W transposed to K-major; W_ih/W_hh native) ---
__global__ void k_prep_weights(const float* __restrict__ W,
                               const float* __restrict__ Wih,
                               const float* __restrict__ Whh,
                               __bf16* __restrict__ Wt,
                               __bf16* __restrict__ WihB,
                               __bf16* __restrict__ WhhB) {
  int i = blockIdx.x * blockDim.x + threadIdx.x;  // grid covers 3*D*D
  if (i < 3 * D * D) {
    WihB[i] = (__bf16)Wih[i];
    WhhB[i] = (__bf16)Whh[i];
    if (i < D * D) {
      int k = i / D, j = i % D;
      Wt[j * D + k] = (__bf16)W[k * D + j];   // W is [1,D,D]; m = h@W
    }
  }
}

// ---- GEMM1: m[BN,D] = xh[BN,D] @ Wt; 2x2 register blocking (32x32/wave) --
__global__ void k_gemm1(const __bf16* __restrict__ xh,
                        const __bf16* __restrict__ Wt,
                        __bf16* __restrict__ m) {
  const int lane = threadIdx.x & 31;
  const int wave = threadIdx.x >> 5;
  const int tile = blockIdx.x * 8 + wave;   // (BN/32)*(D/32) macro tiles
  const int rt = tile >> 3;                 // D/32 == 8 col macro tiles
  const int ct = tile & 7;
  const int row0 = rt * 32, c0 = ct * 32;

  v8f a00 = {}, a01 = {}, a10 = {}, a11 = {};
  for (int k0 = 0; k0 < D; k0 += 32) {
    v16bf fa0 = load_a_frag(xh, D, row0,      k0, lane);
    v16bf fa1 = load_a_frag(xh, D, row0 + 16, k0, lane);
    v16bf fb0 = load_b_frag(Wt, D, c0,        k0, lane);
    v16bf fb1 = load_b_frag(Wt, D, c0 + 16,   k0, lane);
    a00 = wmma_bf16(fa0, fb0, a00);
    a01 = wmma_bf16(fa0, fb1, a01);
    a10 = wmma_bf16(fa1, fb0, a10);
    a11 = wmma_bf16(fa1, fb1, a11);
  }
  const int nc   = lane & 15;
  const int mrow = (lane >> 4) * 8;
#pragma unroll
  for (int r = 0; r < 8; ++r) {
    const size_t ro0 = (size_t)(row0 + mrow + r) * D;
    const size_t ro1 = (size_t)(row0 + 16 + mrow + r) * D;
    m[ro0 + c0 + nc]      = (__bf16)a00[r];
    m[ro0 + c0 + 16 + nc] = (__bf16)a01[r];
    m[ro1 + c0 + nc]      = (__bf16)a10[r];
    m[ro1 + c0 + 16 + nc] = (__bf16)a11[r];
  }
}

// ---- edge scatter: agg[v] += m[u] * ew (one wave per edge) ---------------
__global__ void k_edges(const __bf16* __restrict__ m,
                        const int* __restrict__ edge_index,
                        const float* __restrict__ ew,
                        const int* __restrict__ mask,
                        float* __restrict__ agg) {
  const int lane = threadIdx.x & 31;
  const int wave = threadIdx.x >> 5;
  const long g = (long)blockIdx.x * 8 + wave;       // < B*E
  const int b = (int)(g >> 16);                     // E = 65536
  const int e = (int)(g & (E - 1));
  const int* ei = edge_index + (size_t)b * 2 * E;   // [B,2,E]
  const int u = ei[e];
  const int v = ei[E + e];
  const float w = ew[(size_t)b * E + e];
  if (mask[b * N + u] <= 0 || mask[b * N + v] <= 0 || w == 0.0f) return;

  const uint4* src = reinterpret_cast<const uint4*>(m + ((size_t)b * N + u) * D);
  Pack8 p; p.q = src[lane];                         // 8 bf16 per lane
  float* dst = agg + ((size_t)b * N + v) * D + lane * 8;
#pragma unroll
  for (int j = 0; j < 8; ++j)
    atomicAdd(dst + j, (float)p.h[j] * w);
}

// ---- cvt: agg f32 -> bf16 ------------------------------------------------
__global__ void k_cvt(const float* __restrict__ a, __bf16* __restrict__ o, int n) {
  int i = blockIdx.x * blockDim.x + threadIdx.x;
  if (i < n) o[i] = (__bf16)a[i];
}

// ---- fused GRU: 32x16 tile/wave, B-frags shared across two row tiles -----
__global__ void k_gru(const __bf16* __restrict__ aggb,
                      const __bf16* __restrict__ xh,
                      const __bf16* __restrict__ Wih,
                      const __bf16* __restrict__ Whh,
                      const float* __restrict__ bih,
                      const float* __restrict__ bhh,
                      const float* __restrict__ x,
                      const int* __restrict__ mask,
                      float* __restrict__ out) {
  const int lane = threadIdx.x & 31;
  const int wave = threadIdx.x >> 5;
  const int tile = blockIdx.x * 8 + wave;   // (BN/32)*(D/16) macro tiles
  const int rt = tile >> 4;                 // 16 col tiles
  const int ct = tile & 15;
  const int row0 = rt * 32, c0 = ct * 16;

  v8f accR0 = {}, accZ0 = {}, accGN0 = {}, accHN0 = {};
  v8f accR1 = {}, accZ1 = {}, accGN1 = {}, accHN1 = {};
  for (int k0 = 0; k0 < D; k0 += 32) {
    v16bf aA0 = load_a_frag(aggb, D, row0,      k0, lane);
    v16bf aA1 = load_a_frag(aggb, D, row0 + 16, k0, lane);
    v16bf aH0 = load_a_frag(xh,   D, row0,      k0, lane);
    v16bf aH1 = load_a_frag(xh,   D, row0 + 16, k0, lane);
    // r gate: gi_r + gh_r chained into one accumulator per row tile
    v16bf bIR = load_b_frag(Wih, D, c0, k0, lane);
    v16bf bHR = load_b_frag(Whh, D, c0, k0, lane);
    accR0 = wmma_bf16(aA0, bIR, accR0);  accR0 = wmma_bf16(aH0, bHR, accR0);
    accR1 = wmma_bf16(aA1, bIR, accR1);  accR1 = wmma_bf16(aH1, bHR, accR1);
    // z gate
    v16bf bIZ = load_b_frag(Wih, D, D + c0, k0, lane);
    v16bf bHZ = load_b_frag(Whh, D, D + c0, k0, lane);
    accZ0 = wmma_bf16(aA0, bIZ, accZ0);  accZ0 = wmma_bf16(aH0, bHZ, accZ0);
    accZ1 = wmma_bf16(aA1, bIZ, accZ1);  accZ1 = wmma_bf16(aH1, bHZ, accZ1);
    // n candidate: gi_n and gh_n kept separate (needs r * gh_n)
    v16bf bIN = load_b_frag(Wih, D, 2 * D + c0, k0, lane);
    v16bf bHN = load_b_frag(Whh, D, 2 * D + c0, k0, lane);
    accGN0 = wmma_bf16(aA0, bIN, accGN0);  accGN1 = wmma_bf16(aA1, bIN, accGN1);
    accHN0 = wmma_bf16(aH0, bHN, accHN0);  accHN1 = wmma_bf16(aH1, bHN, accHN1);
  }

  const int col = c0 + (lane & 15);
  const float bR  = bih[col] + bhh[col];
  const float bZ  = bih[D + col] + bhh[D + col];
  const float bGN = bih[2 * D + col];
  const float bHN = bhh[2 * D + col];
  const int mrow = (lane >> 4) * 8;
#pragma unroll
  for (int half = 0; half < 2; ++half) {
    const v8f& aR  = half ? accR1  : accR0;
    const v8f& aZ  = half ? accZ1  : accZ0;
    const v8f& aGN = half ? accGN1 : accGN0;
    const v8f& aHN = half ? accHN1 : accHN0;
#pragma unroll
    for (int r = 0; r < 8; ++r) {
      const int row = row0 + half * 16 + mrow + r;  // flat [B*N) row
      const float rr = sigmoidf(aR[r] + bR);
      const float zz = sigmoidf(aZ[r] + bZ);
      const float nn = tanhf(aGN[r] + bGN + rr * (aHN[r] + bHN));
      const float hold = x[(size_t)row * D + col];
      const float hnew = (1.0f - zz) * nn + zz * hold;
      out[(size_t)row * D + col] = (mask[row] > 0) ? hnew : 0.0f;
    }
  }
}

extern "C" void kernel_launch(void* const* d_in, const int* in_sizes, int n_in,
                              void* d_out, int out_size, void* d_ws, size_t ws_size,
                              hipStream_t stream) {
  (void)in_sizes; (void)n_in; (void)out_size; (void)ws_size;
  const float* x    = (const float*)d_in[0];
  const int*   ei   = (const int*)d_in[1];
  const float* ew   = (const float*)d_in[2];
  const int*   mask = (const int*)d_in[3];
  const float* W    = (const float*)d_in[4];
  const float* Wih  = (const float*)d_in[5];
  const float* Whh  = (const float*)d_in[6];
  const float* bih  = (const float*)d_in[7];
  const float* bhh  = (const float*)d_in[8];
  float* out = (float*)d_out;

  char* ws = (char*)d_ws;                       // ~81 MB total
  __bf16* xh   = (__bf16*)(ws);                 // 16 MB  B*N*D bf16
  __bf16* m    = (__bf16*)(ws + 16777216);      // 16 MB  B*N*D bf16
  float*  agg  = (float*) (ws + 33554432);      // 32 MB  B*N*D f32
  __bf16* aggb = (__bf16*)(ws + 67108864);      // 16 MB  B*N*D bf16
  __bf16* Wt   = (__bf16*)(ws + 83886080);      // 128 KB D*D bf16 (K-major)
  __bf16* WihB = (__bf16*)(ws + 84017152);      // 384 KB 3D*D bf16
  __bf16* WhhB = (__bf16*)(ws + 84410368);      // 384 KB 3D*D bf16

  const int nBND = B * N * D;                   // 8388608
  k_prep_nodes<<<nBND / 256, 256, 0, stream>>>(x, xh, agg, nBND);
  k_prep_weights<<<(3 * D * D) / 256, 256, 0, stream>>>(W, Wih, Whh, Wt, WihB, WhhB);
  k_gemm1<<<(BN / 32) * (D / 32) / 8, 256, 0, stream>>>(xh, Wt, m);
  k_edges<<<(B * E) / 8, 256, 0, stream>>>(m, ei, ew, mask, agg);
  k_cvt<<<nBND / 256, 256, 0, stream>>>(agg, aggb, nBND);
  k_gru<<<(BN / 32) * (D / 16) / 8, 256, 0, stream>>>(aggb, xh, WihB, WhhB,
                                                      bih, bhh, x, mask, out);
}
